// Net_88407606821102
// MI455X (gfx1250) — matrix operand order repaired and checked by reading, not compiled, not run on hardware
//
#include <hip/hip_runtime.h>

#define N_NODES   10000
#define E_EDGES   160000
#define ET_EDGES  170000     // E + N self loops
#define FDIM      768
#define HHEADS    8
#define DDIM      96
#define NLAYERS   4
#define NEG_SLOPE 0.2f
#define HEAD_NEG  0.01f

#define BM 128
#define BN 64
#define BK 32
#define LSTR 40   // bf16 elements per LDS row (32 + 8 pad -> 20 dwords, conflict-free)
#define M_PAD 10112  // ceil(10000/128)*128 -> unconditional tile staging

typedef __attribute__((ext_vector_type(16))) __bf16 v16bf;
typedef __attribute__((ext_vector_type(8)))  __bf16 bf16x8;
typedef __attribute__((ext_vector_type(8)))  float  v8f;

// ---------------------------------------------------------------- conversions

// fp32 -> bf16, 8 elements/thread; rows beyond n8_valid are zero padding
__global__ void k_f32_to_bf16_vec8(const float* __restrict__ in,
                                   __bf16* __restrict__ out,
                                   int n8_valid, int n8_total) {
  int i = blockIdx.x * blockDim.x + threadIdx.x;
  if (i >= n8_total) return;
  bf16x8 r = {};
  if (i < n8_valid) {
    const float4* p = (const float4*)in + (size_t)i * 2;
    float4 a = p[0], b = p[1];
    r[0] = (__bf16)a.x; r[1] = (__bf16)a.y; r[2] = (__bf16)a.z; r[3] = (__bf16)a.w;
    r[4] = (__bf16)b.x; r[5] = (__bf16)b.y; r[6] = (__bf16)b.z; r[7] = (__bf16)b.w;
  }
  ((bf16x8*)out)[i] = r;
}

// W[l][k][n] (fp32) -> WT[l][n][k] (bf16), N-major for the WMMA B operand
__global__ void k_transpose_to_bf16(const float* __restrict__ W,
                                    __bf16* __restrict__ WT, int total) {
  int i = blockIdx.x * blockDim.x + threadIdx.x;
  if (i >= total) return;
  int l = i / (FDIM * FDIM);
  int rem = i - l * FDIM * FDIM;
  int n = rem / FDIM;
  int k = rem - n * FDIM;
  WT[i] = (__bf16)W[(size_t)l * FDIM * FDIM + (size_t)k * FDIM + n];
}

// ---------------------------------------------------------------- WMMA GEMM
// C[M x 768] = A[M_PAD x 768](bf16) * BT[768 x 768]^T(bf16, N-major) + bias
// Async global->LDS staging (ASYNCcnt) + double-buffered LDS.

__global__ __launch_bounds__(256) void k_gemm_bf16_wmma(
    const __bf16* __restrict__ A, const __bf16* __restrict__ BT,
    const float* __restrict__ bias, float* __restrict__ C, int M) {
  __shared__ __bf16 lA[2][BM * LSTR];
  __shared__ __bf16 lB[2][BN * LSTR];
  const int tid  = threadIdx.x;
  const int lane = tid & 31;
  const int wave = tid >> 5;          // 0..7
  const int wm   = (wave >> 1) * 32;  // wave M offset inside block tile
  const int wn   = (wave & 1) * 32;   // wave N offset
  const int half = lane >> 4;
  const int mr   = lane & 15;
  const int blockM = blockIdx.y * BM;
  const int blockN = blockIdx.x * BN;

  // per-thread staging coordinates (16B chunks)
  const int rowA0 = tid >> 2;           // chunk tid        -> rows 0..63
  const int rowA1 = (tid + 256) >> 2;   // chunk tid+256    -> rows 64..127
  const int cc    = (tid & 3) * 8;      // bf16 column offset within 32-wide tile

  const __bf16* gA0 = A + (size_t)(blockM + rowA0) * FDIM + cc;
  const __bf16* gA1 = A + (size_t)(blockM + rowA1) * FDIM + cc;
  const __bf16* gB  = BT + (size_t)(blockN + rowA0) * FDIM + cc;

  v8f acc[2][2] = {};

  auto stage = [&](int kk, int b) {
    unsigned la0 = (unsigned)(unsigned long long)(const void*)&lA[b][rowA0 * LSTR + cc];
    unsigned la1 = (unsigned)(unsigned long long)(const void*)&lA[b][rowA1 * LSTR + cc];
    unsigned lb  = (unsigned)(unsigned long long)(const void*)&lB[b][rowA0 * LSTR + cc];
    asm volatile("global_load_async_to_lds_b128 %0, %1, off"
                 :: "v"(la0), "v"((unsigned long long)(gA0 + kk)) : "memory");
    asm volatile("global_load_async_to_lds_b128 %0, %1, off"
                 :: "v"(la1), "v"((unsigned long long)(gA1 + kk)) : "memory");
    asm volatile("global_load_async_to_lds_b128 %0, %1, off"
                 :: "v"(lb),  "v"((unsigned long long)(gB + kk)) : "memory");
  };

  stage(0, 0);
  int buf = 0;
  for (int kk = 0; kk < FDIM; kk += BK) {
    asm volatile("s_wait_asynccnt 0x0" ::: "memory");  // own tile chunks resident
    __syncthreads();                                   // everyone's chunks resident
    if (kk + BK < FDIM) stage(kk + BK, buf ^ 1);       // overlap next fill with math

    // A fragment: lane holds row M=mr, K = {half*8..+7} U {16+half*8..+7}
    v16bf afrag[2], bfrag[2];
#pragma unroll
    for (int i = 0; i < 2; ++i) {
      const __bf16* pa = &lA[buf][(wm + i * 16 + mr) * LSTR];
      bf16x8 lo = *(const bf16x8*)(pa + half * 8);
      bf16x8 hi = *(const bf16x8*)(pa + 16 + half * 8);
#pragma unroll
      for (int t = 0; t < 8; ++t) { afrag[i][t] = lo[t]; afrag[i][t + 8] = hi[t]; }
    }
    // B fragment: lane holds col N=mr, K = {half*8..+7} U {16+half*8..+7}
#pragma unroll
    for (int j = 0; j < 2; ++j) {
      const __bf16* pb = &lB[buf][(wn + j * 16 + mr) * LSTR] + half * 8;
      bf16x8 lo = *(const bf16x8*)(pb);
      bf16x8 hi = *(const bf16x8*)(pb + 16);
#pragma unroll
      for (int t = 0; t < 8; ++t) { bfrag[j][t] = lo[t]; bfrag[j][t + 8] = hi[t]; }
    }
#pragma unroll
    for (int i = 0; i < 2; ++i)
#pragma unroll
      for (int j = 0; j < 2; ++j)
        acc[i][j] = __builtin_amdgcn_wmma_f32_16x16x32_bf16(
            false, afrag[i], false, bfrag[j], (short)0, acc[i][j], false, false);
    buf ^= 1;
  }

  // C layout: VGPR r -> M = r + half*8, N = mr
#pragma unroll
  for (int i = 0; i < 2; ++i) {
#pragma unroll
    for (int j = 0; j < 2; ++j) {
      int col = blockN + wn + j * 16 + mr;
      float bv = bias[col];
#pragma unroll
      for (int r = 0; r < 8; ++r) {
        int row = blockM + wm + i * 16 + half * 8 + r;
        if (row < M) C[(size_t)row * FDIM + col] = acc[i][j][r] + bv;
      }
    }
  }
}

// ---------------------------------------------------------------- CSR build

__global__ void k_zero_i32(int* p, int n) {
  int i = blockIdx.x * blockDim.x + threadIdx.x;
  if (i < n) p[i] = 0;
}

__global__ void k_count_deg(const int* __restrict__ dst, int* __restrict__ deg) {
  int e = blockIdx.x * blockDim.x + threadIdx.x;
  if (e >= ET_EDGES) return;
  int t = (e < E_EDGES) ? dst[e] : (e - E_EDGES);
  atomicAdd(&deg[t], 1);
}

__global__ __launch_bounds__(1024) void k_scan_rowptr(const int* __restrict__ deg,
                                                      int* __restrict__ rowptr,
                                                      int* __restrict__ fill) {
  __shared__ int sums[1024];
  int t = threadIdx.x;
  const int CH = (N_NODES + 1023) / 1024;
  int base = t * CH;
  int s = 0;
  for (int i = 0; i < CH; ++i) {
    int idx = base + i;
    if (idx < N_NODES) s += deg[idx];
  }
  sums[t] = s;
  __syncthreads();
  for (int o = 1; o < 1024; o <<= 1) {
    int v = (t >= o) ? sums[t - o] : 0;
    __syncthreads();
    sums[t] += v;
    __syncthreads();
  }
  int run = (t == 0) ? 0 : sums[t - 1];
  for (int i = 0; i < CH; ++i) {
    int idx = base + i;
    if (idx < N_NODES) {
      rowptr[idx] = run;
      fill[idx] = run;
      run += deg[idx];
    }
  }
  if (t == 1023) rowptr[N_NODES] = sums[1023];
}

__global__ void k_scatter_edges(const int* __restrict__ src, const int* __restrict__ dst,
                                int* __restrict__ fill, int* __restrict__ eidx,
                                int* __restrict__ esrcv) {
  int e = blockIdx.x * blockDim.x + threadIdx.x;
  if (e >= ET_EDGES) return;
  int s = (e < E_EDGES) ? src[e] : (e - E_EDGES);
  int t = (e < E_EDGES) ? dst[e] : (e - E_EDGES);
  int pos = atomicAdd(&fill[t], 1);
  eidx[pos] = e;
  esrcv[pos] = s;
}

// ---------------------------------------------------------------- edge phase

__global__ void k_edge_logits(const float* __restrict__ fs, const float* __restrict__ fd,
                              const float* __restrict__ attn, const int* __restrict__ src,
                              const int* __restrict__ dst, float* __restrict__ logits) {
  int id = blockIdx.x * blockDim.x + threadIdx.x;
  if (id >= ET_EDGES * HHEADS) return;
  int e = id >> 3;
  int h = id & 7;
  int s = (e < E_EDGES) ? src[e] : (e - E_EDGES);
  int t = (e < E_EDGES) ? dst[e] : (e - E_EDGES);
  const float4* ps = (const float4*)(fs + (size_t)s * FDIM + h * DDIM);
  const float4* pd = (const float4*)(fd + (size_t)t * FDIM + h * DDIM);
  const float4* pa = (const float4*)(attn + h * DDIM);
  float acc = 0.f;
#pragma unroll
  for (int i = 0; i < DDIM / 4; ++i) {
    float4 a = ps[i], b = pd[i], w = pa[i];
    float x;
    x = a.x + b.x; x = x > 0.f ? x : NEG_SLOPE * x; acc += w.x * x;
    x = a.y + b.y; x = x > 0.f ? x : NEG_SLOPE * x; acc += w.y * x;
    x = a.z + b.z; x = x > 0.f ? x : NEG_SLOPE * x; acc += w.z * x;
    x = a.w + b.w; x = x > 0.f ? x : NEG_SLOPE * x; acc += w.w * x;
  }
  logits[id] = acc;
}

// one wave32 per (node, head): segment softmax + weighted aggregation
__global__ __launch_bounds__(256) void k_node_softmax_aggregate(
    const int* __restrict__ rowptr, const int* __restrict__ eidx,
    const int* __restrict__ esrcv, const float* __restrict__ logits,
    const float* __restrict__ fs, const float* __restrict__ gbias,
    float* __restrict__ out) {
  int wid = blockIdx.x * (blockDim.x >> 5) + (threadIdx.x >> 5);
  int lane = threadIdx.x & 31;
  if (wid >= N_NODES * HHEADS) return;
  int n = wid >> 3;
  int h = wid & 7;
  int beg = rowptr[n], end = rowptr[n + 1];

  float m = -3.4e38f;
  for (int j = beg + lane; j < end; j += 32) m = fmaxf(m, logits[eidx[j] * HHEADS + h]);
#pragma unroll
  for (int o = 16; o > 0; o >>= 1) m = fmaxf(m, __shfl_xor(m, o, 32));

  float sum = 0.f;
  for (int j = beg + lane; j < end; j += 32) sum += __expf(logits[eidx[j] * HHEADS + h] - m);
#pragma unroll
  for (int o = 16; o > 0; o >>= 1) sum += __shfl_xor(sum, o, 32);
  float inv = 1.0f / sum;

  float a0 = 0.f, a1 = 0.f, a2 = 0.f;
  for (int j = beg; j < end; ++j) {
    float a = __expf(logits[eidx[j] * HHEADS + h] - m) * inv;
    const float* ps = fs + (size_t)esrcv[j] * FDIM + h * DDIM;
    a0 += a * ps[lane];
    a1 += a * ps[lane + 32];
    a2 += a * ps[lane + 64];
  }
  float* po = out + (size_t)n * FDIM + h * DDIM;
  const float* pb = gbias + h * DDIM;
  po[lane]      = a0 + pb[lane];
  po[lane + 32] = a1 + pb[lane + 32];
  po[lane + 64] = a2 + pb[lane + 64];
}

// ---------------------------------------------------------------- MLP head

__global__ void k_head_fc1(const float* __restrict__ h, const float* __restrict__ w,
                           const float* __restrict__ b, float* __restrict__ z) {
  int j = blockIdx.x * blockDim.x + threadIdx.x;
  if (j >= 600) return;
  const float4* ph = (const float4*)h;  // node 0 row
  const float4* pw = (const float4*)(w + (size_t)j * FDIM);
  float acc = 0.f;
#pragma unroll 8
  for (int i = 0; i < FDIM / 4; ++i) {
    float4 a = ph[i], c = pw[i];
    acc += a.x * c.x + a.y * c.y + a.z * c.z + a.w * c.w;
  }
  acc += b[j];
  z[j] = acc > 0.f ? acc : HEAD_NEG * acc;
}

__global__ void k_head_fc2(const float* __restrict__ z, const float* __restrict__ w,
                           const float* __restrict__ b, float* __restrict__ y) {
  __shared__ float red[256];
  int t = threadIdx.x;
  float acc = 0.f;
  for (int j = t; j < 600; j += 256) acc += z[j] * w[j];
  red[t] = acc;
  __syncthreads();
  for (int o = 128; o > 0; o >>= 1) {
    if (t < o) red[t] += red[t + o];
    __syncthreads();
  }
  if (t == 0) y[0] = red[0] + b[0];
}

// ---------------------------------------------------------------- launch

extern "C" void kernel_launch(void* const* d_in, const int* in_sizes, int n_in,
                              void* d_out, int out_size, void* d_ws, size_t ws_size,
                              hipStream_t stream) {
  (void)in_sizes; (void)n_in; (void)out_size; (void)ws_size;
  const float* features = (const float*)d_in[0];
  const int*   src      = (const int*)d_in[1];
  const int*   dst      = (const int*)d_in[2];
  const float* Wsrc     = (const float*)d_in[3];
  const float* bsrc     = (const float*)d_in[4];
  const float* Wdst     = (const float*)d_in[5];
  const float* bdst     = (const float*)d_in[6];
  const float* attn     = (const float*)d_in[7];
  const float* gbias    = (const float*)d_in[8];
  const float* fc1_w    = (const float*)d_in[9];
  const float* fc1_b    = (const float*)d_in[10];
  const float* fc2_w    = (const float*)d_in[11];
  const float* fc2_b    = (const float*)d_in[12];
  float* out = (float*)d_out;

  char* base = (char*)d_ws;
  size_t off = 0;
  auto take = [&](size_t bytes) {
    char* r = base + off;
    off = (off + bytes + 255) & ~(size_t)255;
    return r;
  };
  const size_t WT_BYTES = (size_t)NLAYERS * FDIM * FDIM * 2;
  const size_t HBUF     = (size_t)N_NODES * FDIM * 4;
  const size_t HBF_PAD  = (size_t)M_PAD * FDIM * 2;
  const size_t LOGB     = (size_t)ET_EDGES * HHEADS * 4;

  __bf16* wsrcT = (__bf16*)take(WT_BYTES);
  __bf16* wdstT = (__bf16*)take(WT_BYTES);
  float*  h     = (float*) take(HBUF);
  __bf16* hbf   = (__bf16*)take(HBF_PAD);
  float*  fsb   = (float*) take(HBUF);
  float*  fdb   = (float*) take(HBUF);
  float*  logit = (float*) take(LOGB);
  int* deg      = (int*)take((size_t)N_NODES * 4);
  int* rowptr   = (int*)take((size_t)(N_NODES + 1) * 4);
  int* fill     = (int*)take((size_t)N_NODES * 4);
  int* eidx     = (int*)take((size_t)ET_EDGES * 4);
  int* esrcv    = (int*)take((size_t)ET_EDGES * 4);

  // weights -> bf16, transposed to [n][k] (B operand wants K-contiguous per N)
  int wtot = NLAYERS * FDIM * FDIM;
  k_transpose_to_bf16<<<(wtot + 255) / 256, 256, 0, stream>>>(Wsrc, wsrcT, wtot);
  k_transpose_to_bf16<<<(wtot + 255) / 256, 256, 0, stream>>>(Wdst, wdstT, wtot);

  // CSR of incoming edges (built fresh every call)
  k_zero_i32<<<(N_NODES + 255) / 256, 256, 0, stream>>>(deg, N_NODES);
  k_count_deg<<<(ET_EDGES + 255) / 256, 256, 0, stream>>>(dst, deg);
  k_scan_rowptr<<<1, 1024, 0, stream>>>(deg, rowptr, fill);
  k_scatter_edges<<<(ET_EDGES + 255) / 256, 256, 0, stream>>>(src, dst, fill, eidx, esrcv);

  dim3 ggrid(FDIM / BN, M_PAD / BM);
  int n8v = N_NODES * FDIM / 8;
  int n8t = M_PAD * FDIM / 8;
  for (int l = 0; l < NLAYERS; ++l) {
    const float* hin = (l == 0) ? features : h;
    k_f32_to_bf16_vec8<<<(n8t + 255) / 256, 256, 0, stream>>>(hin, hbf, n8v, n8t);
    k_gemm_bf16_wmma<<<ggrid, 256, 0, stream>>>(
        hbf, wsrcT + (size_t)l * FDIM * FDIM, bsrc + l * FDIM, fsb, N_NODES);
    k_gemm_bf16_wmma<<<ggrid, 256, 0, stream>>>(
        hbf, wdstT + (size_t)l * FDIM * FDIM, bdst + l * FDIM, fdb, N_NODES);
    k_edge_logits<<<(ET_EDGES * HHEADS + 255) / 256, 256, 0, stream>>>(
        fsb, fdb, attn + l * FDIM, src, dst, logit);
    k_node_softmax_aggregate<<<(N_NODES * HHEADS + 7) / 8, 256, 0, stream>>>(
        rowptr, eidx, esrcv, logit, fsb, gbias + l * FDIM, h);
  }

  k_head_fc1<<<3, 256, 0, stream>>>(h, fc1_w, fc1_b, out);         // z -> out[0..599]
  k_head_fc2<<<1, 256, 0, stream>>>(out, fc2_w, fc2_b, out + 600); // y -> out[600]
}
